// Chevby_GCN_77627238908519
// MI455X (gfx1250) — matrix operand order repaired and checked by reading, not compiled, not run on hardware
//
#include <hip/hip_runtime.h>

// ---------------------------------------------------------------------------
// Chebyshev GCN, fused, bf16 WMMA (gfx1250 / MI455X)
//   out[b,n,o,t] = relu( sum_k sum_c theta[k,c,o] *
//                        sum_m (cheb[k,m,n]*STDG[b,m,n]) * x[b,m,c,t] )
// One WG = (b, 64 n-rows, 4 t-cols). Phase 1: rhs_k = masked^T @ X via
// v_wmma_f32_16x16x32_bf16, all 3 k accumulated in registers. Phase 2:
// rhs_k -> LDS (bf16), GEMM with theta_k, ReLU, store.
// ---------------------------------------------------------------------------

typedef __attribute__((ext_vector_type(8)))  float  v8f;
typedef __attribute__((ext_vector_type(16))) __bf16 v16bf;
typedef __attribute__((ext_vector_type(8)))  __bf16 v8bf;

#define DEVI static __device__ __forceinline__

constexpr int B_  = 16, N_ = 1024, CI = 64, T_ = 24, KK = 3, CO = 64;
constexpr int BT  = 4;            // t-block per WG
constexpr int BM  = 64;           // n-rows per WG
constexpr int MK  = 32;           // m-chunk (= WMMA K for bf16)
constexpr int JC  = CI * BT;      // phase-1 column count (c,t) = 256
constexpr int AP  = 40;           // ldsA pitch in bf16 (80B: 16B-aligned rows)
constexpr int XP  = 40;           // ldsX pitch in bf16
constexpr int RP  = 72;           // ldsRhs / ldsTheta pitch in bf16 (144B)

constexpr int LDSA_B = KK * BM * AP * 2;   // 15360
constexpr int LDSX_B = JC * XP * 2;        // 20480
constexpr int LDSR_B = BT * BM * RP * 2;   // 36864
constexpr int LDST_B = CO * RP * 2;        //  9216
constexpr int SMEM_B = (LDSA_B + LDSX_B) > (LDSR_B + LDST_B)
                         ? (LDSA_B + LDSX_B) : (LDSR_B + LDST_B); // 46080

DEVI unsigned pack_bf16(float a, float b) {
  __bf16 ha = (__bf16)a, hb = (__bf16)b;
  unsigned short ua, ub;
  __builtin_memcpy(&ua, &ha, 2);
  __builtin_memcpy(&ub, &hb, 2);
  return (unsigned)ua | ((unsigned)ub << 16);
}

DEVI v16bf cat8(v8bf a, v8bf b) {
  return __builtin_shufflevector(a, b, 0,1,2,3,4,5,6,7,8,9,10,11,12,13,14,15);
}

__global__ __launch_bounds__(256, 1)
void cheb_gcn_fused(const float* __restrict__ x, const float* __restrict__ STDG,
                    const float* __restrict__ cheb, const float* __restrict__ theta,
                    float* __restrict__ out)
{
  __shared__ __align__(16) unsigned char smem[SMEM_B];
  __bf16* ldsA = (__bf16*)(smem);            // [k][n(64)][m(32)]  pitch AP
  __bf16* ldsX = (__bf16*)(smem + LDSA_B);   // [j(256)][m(32)]    pitch XP, j=tt*64+c
  __bf16* ldsR = (__bf16*)(smem);            // [tt(4)][n(64)][c]  pitch RP (reuses region)
  __bf16* ldsT = (__bf16*)(smem + LDSR_B);   // [o(64)][c(64)]     pitch RP

  const int tid  = threadIdx.x;
  const int w    = tid >> 5;
  const int lane = tid & 31;
  const int l16  = lane & 15;
  const int lhi  = lane >> 4;          // 0 or 1

  const int b  = blockIdx.z;
  const int nb = blockIdx.y * BM;
  const int t0 = blockIdx.x * BT;

  // wave tiling: 2 n-tiles x 4 column-tiles per wave (4x16 tile grid total)
  const int tnA = 2 * (w & 1);
  const int jB  = 4 * (w >> 1);        // also the phase-2 slot base

  v8f acc1[KK][2][4];                  // rhs_k accumulators (f32)
#pragma unroll
  for (int k = 0; k < KK; ++k)
#pragma unroll
    for (int tn = 0; tn < 2; ++tn)
#pragma unroll
      for (int jt = 0; jt < 4; ++jt)
#pragma unroll
        for (int e = 0; e < 8; ++e) acc1[k][tn][jt][e] = 0.0f;

  // =========================== Phase 1: rhs_k ===============================
  for (int m0 = 0; m0 < N_; m0 += MK) {
    __syncthreads();   // previous chunk fully consumed before re-staging

    // --- stage masked^T (bf16) for all 3 k: ldsA[k][n][m] = cheb*STDG -------
    {
      const int nn = tid & 63, mg = tid >> 6;
#pragma unroll
      for (int i = 0; i < 8; ++i) {
        const int m   = mg * 8 + i;
        const int off = (m0 + m) * N_ + nb + nn;
        const float s = STDG[b * (N_ * N_) + off];
#pragma unroll
        for (int k = 0; k < KK; ++k)
          ldsA[(k * BM + nn) * AP + m] = (__bf16)(cheb[k * (N_ * N_) + off] * s);
      }
    }
    // --- stage X tile transposed: ldsX[j=tt*64+c][m], pairs of m packed -----
    {
      const int c = tid & 63, q = tid >> 6;
#pragma unroll
      for (int p = 0; p < 4; ++p) {
        const int mp = p * 4 + q;                       // m-pair 0..15
        const float* xp = x + (((size_t)(b * N_ + m0 + 2 * mp) * CI + c) * T_ + t0);
        const float4 lo4 = *(const float4*)xp;
        const float4 hi4 = *(const float4*)(xp + CI * T_);
        const float lo[4] = {lo4.x, lo4.y, lo4.z, lo4.w};
        const float hi[4] = {hi4.x, hi4.y, hi4.z, hi4.w};
#pragma unroll
        for (int tt = 0; tt < 4; ++tt)
          *(unsigned*)&ldsX[(tt * CI + c) * XP + 2 * mp] = pack_bf16(lo[tt], hi[tt]);
      }
    }
    __syncthreads();

    // --- B operands (shared across all 3 k): CDNA5 B layout -----------------
    v16bf bB[4];
#pragma unroll
    for (int jt = 0; jt < 4; ++jt) {
      const int j  = (jB + jt) * 16 + l16;   // lane = column
      const int mo = lhi * 8;                // lanes 0-15: K{0..7,16..23}, 16-31: K{8..15,24..31}
      v8bf p0 = *(const v8bf*)&ldsX[j * XP + mo];
      v8bf p1 = *(const v8bf*)&ldsX[j * XP + mo + 16];
      bB[jt] = cat8(p0, p1);
    }
    // --- WMMAs: 3k x 2 n-tiles x 4 col-tiles = 24 per wave per chunk --------
#pragma unroll
    for (int k = 0; k < KK; ++k) {
#pragma unroll
      for (int tn = 0; tn < 2; ++tn) {
        const int n  = (tnA + tn) * 16 + l16;  // lane = row
        const int mo = lhi * 8;                // A layout: {0..7,16..23}/{8..15,24..31}
        v8bf a0 = *(const v8bf*)&ldsA[(k * BM + n) * AP + mo];
        v8bf a1 = *(const v8bf*)&ldsA[(k * BM + n) * AP + mo + 16];
        v16bf aA = cat8(a0, a1);
#pragma unroll
        for (int jt = 0; jt < 4; ++jt)
          acc1[k][tn][jt] = __builtin_amdgcn_wmma_f32_16x16x32_bf16(
              false, aA, false, bB[jt], (short)0, acc1[k][tn][jt], false, false);
      }
    }
  }

  // =========================== Phase 2: theta + ReLU ========================
  v8f acc2[2][4];
#pragma unroll
  for (int tn = 0; tn < 2; ++tn)
#pragma unroll
    for (int s = 0; s < 4; ++s)
#pragma unroll
      for (int e = 0; e < 8; ++e) acc2[tn][s][e] = 0.0f;

#pragma unroll
  for (int k = 0; k < KK; ++k) {
    __syncthreads();   // ldsR/ldsT region free (overlaps ldsA/ldsX)

    // stage theta_k transposed: ldsT[o][c] = theta[k,c,o] (bf16)
    {
      const int o = tid & 63, cb = (tid >> 6) * 16;
#pragma unroll
      for (int i = 0; i < 16; ++i)
        ldsT[o * RP + cb + i] = (__bf16)theta[(size_t)(k * CI + cb + i) * CO + o];
    }
    // dump rhs_k (C/D layout -> [tt][n][c] bf16)
#pragma unroll
    for (int tn = 0; tn < 2; ++tn)
#pragma unroll
      for (int jt = 0; jt < 4; ++jt) {
        const int j = (jB + jt) * 16 + l16;
        const int c = j & 63, tt = j >> 6;
#pragma unroll
        for (int g = 0; g < 8; ++g) {
          const int n = (tnA + tn) * 16 + lhi * 8 + g;
          ldsR[(tt * BM + n) * RP + c] = (__bf16)acc1[k][tn][jt][g];
        }
      }
    __syncthreads();

    // out[n, o, t] += sum_c rhs_k[n,c,t] * theta_k[c,o]
#pragma unroll
    for (int s = 0; s < 4; ++s) {
      const int slot = jB + s;            // 0..15 across waves
      const int to = slot & 3, tt = slot >> 2;
#pragma unroll
      for (int c0 = 0; c0 < CI; c0 += 32) {
        const int o = to * 16 + l16;
        v8bf q0 = *(const v8bf*)&ldsT[o * RP + c0 + lhi * 8];
        v8bf q1 = *(const v8bf*)&ldsT[o * RP + c0 + lhi * 8 + 16];
        v16bf bT = cat8(q0, q1);
#pragma unroll
        for (int tn = 0; tn < 2; ++tn) {
          const int n = (tnA + tn) * 16 + l16;
          v8bf a0 = *(const v8bf*)&ldsR[(tt * BM + n) * RP + c0 + lhi * 8];
          v8bf a1 = *(const v8bf*)&ldsR[(tt * BM + n) * RP + c0 + lhi * 8 + 16];
          v16bf aR = cat8(a0, a1);
          acc2[tn][s] = __builtin_amdgcn_wmma_f32_16x16x32_bf16(
              false, aR, false, bT, (short)0, acc2[tn][s], false, false);
        }
      }
    }
  }

  // =========================== Store (ReLU) =================================
#pragma unroll
  for (int tn = 0; tn < 2; ++tn)
#pragma unroll
    for (int s = 0; s < 4; ++s) {
      const int slot = jB + s;
      const int to = slot & 3, tt = slot >> 2;
      const int o = to * 16 + l16, t = t0 + tt;
#pragma unroll
      for (int g = 0; g < 8; ++g) {
        const int n = nb + (tnA + tn) * 16 + lhi * 8 + g;
        out[(((size_t)b * N_ + n) * CO + o) * T_ + t] = fmaxf(acc2[tn][s][g], 0.0f);
      }
    }
}

extern "C" void kernel_launch(void* const* d_in, const int* in_sizes, int n_in,
                              void* d_out, int out_size, void* d_ws, size_t ws_size,
                              hipStream_t stream) {
  (void)in_sizes; (void)n_in; (void)d_ws; (void)ws_size; (void)out_size;
  const float* x     = (const float*)d_in[0];
  const float* STDG  = (const float*)d_in[1];
  const float* cheb  = (const float*)d_in[2];
  const float* theta = (const float*)d_in[3];
  float* out = (float*)d_out;

  dim3 grid(T_ / BT, N_ / BM, B_);   // (6, 16, 16)
  dim3 block(256);                   // 8 wave32 waves
  cheb_gcn_fused<<<grid, block, 0, stream>>>(x, STDG, cheb, theta, out);
}
